// DualGNN_17257178596107
// MI455X (gfx1250) — matrix operand order repaired and checked by reading, not compiled, not run on hardware
//
#include <hip/hip_runtime.h>
#include <hip/hip_bf16.h>

#define N_NODES 50000
#define NUM_MI  20000
#define D_FEAT  128
#define N_EDGES 600000
#define N_PAIRS 200000
#define LN_EPS  1e-5f
#define INV_TEMP (1.0f / 0.7f)

typedef __attribute__((ext_vector_type(16))) __bf16 v16bf;
typedef __attribute__((ext_vector_type(8)))  float  v8f;

union Frag {
    v16bf        b;
    unsigned int w[8];
};

__device__ __forceinline__ unsigned short f2bf(float f) {
    unsigned int u = __builtin_bit_cast(unsigned int, f);
    unsigned int r = u + 0x7FFFu + ((u >> 16) & 1u);   // round-to-nearest-even
    return (unsigned short)(r >> 16);
}
__device__ __forceinline__ unsigned int pack2bf(float lo, float hi) {
    return (unsigned int)f2bf(lo) | ((unsigned int)f2bf(hi) << 16);
}

// ---------------------------------------------------------------------------
// row_ptr[r] = lower_bound(adj_row, r)   (adj_row is sorted)
// ---------------------------------------------------------------------------
__global__ void k_rowptr(const int* __restrict__ rows, int* __restrict__ rp) {
    int r = blockIdx.x * blockDim.x + threadIdx.x;
    if (r > N_NODES) return;
    int lo = 0, hi = N_EDGES;
    while (lo < hi) {
        int mid = (lo + hi) >> 1;
        if (rows[mid] < r) lo = mid + 1; else hi = mid;
    }
    rp[r] = lo;
}

// ---------------------------------------------------------------------------
// X0 = node_emb + type_emb[type]
// ---------------------------------------------------------------------------
__global__ void k_init(const float* __restrict__ emb, const float* __restrict__ temb,
                       float* __restrict__ X) {
    long long i = (long long)blockIdx.x * blockDim.x + threadIdx.x;   // float4 index
    if (i >= (long long)N_NODES * (D_FEAT / 4)) return;
    long long row = i >> 5;
    int c4 = (int)(i & 31);
    const float4 e = *(const float4*)(emb + row * D_FEAT + c4 * 4);
    const float4 t = *(const float4*)(temb + (row < NUM_MI ? 0 : D_FEAT) + c4 * 4);
    float4 o; o.x = e.x + t.x; o.y = e.y + t.y; o.z = e.z + t.z; o.w = e.w + t.w;
    *(float4*)(X + row * D_FEAT + c4 * 4) = o;
}

// ---------------------------------------------------------------------------
// Wave-per-row LayerNorm over D=128 (lane holds 4 contiguous floats)
// ---------------------------------------------------------------------------
__device__ __forceinline__ float4 wave_ln(float4 x, const float* __restrict__ g,
                                          const float* __restrict__ b, int lane) {
    float s  = x.x + x.y + x.z + x.w;
    float sq = fmaf(x.x, x.x, fmaf(x.y, x.y, fmaf(x.z, x.z, x.w * x.w)));
    #pragma unroll
    for (int off = 16; off; off >>= 1) {
        s  += __shfl_xor(s,  off, 32);
        sq += __shfl_xor(sq, off, 32);
    }
    float mean = s * (1.0f / D_FEAT);
    float var  = sq * (1.0f / D_FEAT) - mean * mean;
    float rstd = rsqrtf(var + LN_EPS);
    const float4 gg = *(const float4*)(g + lane * 4);
    const float4 bb = *(const float4*)(b + lane * 4);
    float4 y;
    y.x = (x.x - mean) * rstd * gg.x + bb.x;
    y.y = (x.y - mean) * rstd * gg.y + bb.y;
    y.z = (x.z - mean) * rstd * gg.z + bb.z;
    y.w = (x.w - mean) * rstd * gg.w + bb.w;
    return y;
}

__global__ void k_ln(const float* __restrict__ X, const float* __restrict__ g,
                     const float* __restrict__ b, float* __restrict__ Y) {
    int wave = threadIdx.x >> 5, lane = threadIdx.x & 31;
    int row = blockIdx.x * 8 + wave;
    if (row >= N_NODES) return;
    float4 x = *(const float4*)(X + (size_t)row * D_FEAT + lane * 4);
    *(float4*)(Y + (size_t)row * D_FEAT + lane * 4) = wave_ln(x, g, b, lane);
}

// ---------------------------------------------------------------------------
// out[r] = sum_e val[e]*H[col[e]] + H[r] (self loop); optional relu; + resid
// Deterministic wave-per-row CSR reduction.
// ---------------------------------------------------------------------------
__global__ void k_spmm(const int* __restrict__ rp, const int* __restrict__ cols,
                       const float* __restrict__ vals, const float* __restrict__ H,
                       const float* __restrict__ resid, float* __restrict__ out,
                       int relu) {
    int wave = threadIdx.x >> 5, lane = threadIdx.x & 31;
    int row = blockIdx.x * 8 + wave;
    if (row >= N_NODES) return;
    int e0 = rp[row], e1 = rp[row + 1];
    // self loop (weight 1.0)
    float4 acc = *(const float4*)(H + (size_t)row * D_FEAT + lane * 4);
    for (int e = e0; e < e1; ++e) {
        float w = vals[e];
        int c = cols[e];
        if (e + 1 < e1)
            __builtin_prefetch(H + (size_t)cols[e + 1] * D_FEAT + lane * 4, 0, 0);
        const float4 h = *(const float4*)(H + (size_t)c * D_FEAT + lane * 4);
        acc.x = fmaf(w, h.x, acc.x); acc.y = fmaf(w, h.y, acc.y);
        acc.z = fmaf(w, h.z, acc.z); acc.w = fmaf(w, h.w, acc.w);
    }
    if (relu) {
        acc.x = fmaxf(acc.x, 0.f); acc.y = fmaxf(acc.y, 0.f);
        acc.z = fmaxf(acc.z, 0.f); acc.w = fmaxf(acc.w, 0.f);
    }
    if (resid) {
        const float4 r = *(const float4*)(resid + (size_t)row * D_FEAT + lane * 4);
        acc.x += r.x; acc.y += r.y; acc.z += r.z; acc.w += r.w;
    }
    *(float4*)(out + (size_t)row * D_FEAT + lane * 4) = acc;
}

// ---------------------------------------------------------------------------
// Generic WMMA matmul: C[M x 128] = act(A[M x K] @ W[K x 128] + bias) [+ resid]
//   A from one pointer (K=128), two pointers (concat, K=256), or pair-gathered.
//   If w2 != null: fused final projection out[m] = relu-row . w2 + b2 (mlp path).
// Block = 256 threads = 8 waves; wave computes a 16x128 tile.
// Fully unrolled K-loop + explicit ping-pong double buffers for A and B
// fragments so each WMMA overlaps with the DS loads of the next fragment
// (partial s_wait_dscnt instead of full drains).
// LDS: W as bf16 col-major [128][K] + A block as bf16 row-major [128][K].
// ---------------------------------------------------------------------------
template <int K>
__global__ __launch_bounds__(256)
void k_mm(const float* __restrict__ A0, const float* __restrict__ A1,
          const int* __restrict__ pairs,
          const float* __restrict__ W, const float* __restrict__ bias,
          const float* __restrict__ resid, float* __restrict__ Cout,
          const float* __restrict__ w2, const float* __restrict__ b2p,
          int M, int relu) {
    extern __shared__ unsigned short smem[];
    unsigned short* Wl = smem;                 // [col][k]  col-major, 128*K
    unsigned short* Al = smem + 128 * K;       // [row][k]  row-major, 128*K

    const int tid = threadIdx.x;
    const int rowBase = blockIdx.x * 128;

    // ---- stage W (fp32 -> bf16, transpose to col-major for contiguous-K reads)
    #pragma unroll 4
    for (int idx = tid; idx < K * 128; idx += 256) {
        int k = idx >> 7, col = idx & 127;     // coalesced read of W[k*128+col]
        Wl[col * K + k] = f2bf(W[idx]);
    }
    // ---- stage A block (128 rows x K), fp32 -> bf16, zero-pad past M
    constexpr int lognv = (K == 256) ? 6 : 5;  // float4s per row
    constexpr int nvec = 1 << lognv;
    #pragma unroll 2
    for (int v = tid; v < 128 * nvec; v += 256) {
        int row = v >> lognv;
        int cc = (v & (nvec - 1)) << 2;
        int grow = rowBase + row;
        float4 f = make_float4(0.f, 0.f, 0.f, 0.f);
        if (grow < M) {
            const float* src;
            if (pairs) {
                int u = pairs[2 * grow];
                int vv = pairs[2 * grow + 1] + NUM_MI;
                src = (cc < 128) ? (A0 + (size_t)u * D_FEAT + cc)
                                 : (A0 + (size_t)vv * D_FEAT + (cc - 128));
            } else if (cc >= 128) {
                src = A1 + (size_t)grow * D_FEAT + (cc - 128);
            } else {
                src = A0 + (size_t)grow * D_FEAT + cc;
            }
            f = *(const float4*)src;
        }
        uint2 p; p.x = pack2bf(f.x, f.y); p.y = pack2bf(f.z, f.w);
        *(uint2*)(&Al[row * K + cc]) = p;
    }
    __syncthreads();

    // ---- per-wave 16x128 tile
    const int wave = tid >> 5, lane = tid & 31;
    const int m0 = wave * 16;
    const int half = lane >> 4, mloc = lane & 15;

    v8f acc[8];
    #pragma unroll
    for (int i = 0; i < 8; ++i)
        #pragma unroll
        for (int j = 0; j < 8; ++j) acc[i][j] = 0.f;

    const unsigned short* arow = Al + (m0 + mloc) * K;
    const unsigned short* wbase = Wl + mloc * K + 8 * half;

    // A fragment loader: lane holds row (m0+mloc); element pair t covers
    // K = 16*(t>>2) + 8*half + 2*(t&3) .. +1   (CDNA5 16-bit A 16x32 layout)
    auto loadA = [&](Frag& a, int kt) {
        #pragma unroll
        for (int t = 0; t < 8; ++t) {
            int k0 = ((t & 4) << 2) + 8 * half + ((t & 3) << 1);
            a.w[t] = *(const unsigned int*)(arow + kt + k0);
        }
    };
    // B fragment loader: lane holds col (nt*16+mloc); pair t -> K = 16*(t>>2)+8*half+2*(t&3)
    auto loadB = [&](Frag& b, int nt, int kt) {
        const unsigned short* wcol = wbase + (nt * 16) * K + kt;
        #pragma unroll
        for (int t = 0; t < 8; ++t)
            b.w[t] = *(const unsigned int*)(wcol + ((t & 4) << 2) + ((t & 3) << 1));
    };

    constexpr int KSTEPS = K / 32;
    Frag aa[2], bb[2];
    loadA(aa[0], 0);
    loadB(bb[0], 0, 0);

    #pragma unroll
    for (int kk = 0; kk < KSTEPS; ++kk) {
        const int kt = kk * 32;
        if (kk + 1 < KSTEPS) loadA(aa[(kk + 1) & 1], kt + 32);   // prefetch next A
        #pragma unroll
        for (int nt = 0; nt < 8; ++nt) {
            // prefetch next B fragment while current WMMA executes
            if (nt < 7)                 loadB(bb[(nt + 1) & 1], nt + 1, kt);
            else if (kk + 1 < KSTEPS)   loadB(bb[0], 0, kt + 32);
            acc[nt] = __builtin_amdgcn_wmma_f32_16x16x32_bf16(
                false, aa[kk & 1].b, false, bb[nt & 1].b, (short)0,
                acc[nt], false, false);
        }
    }

    // ---- epilogue
    if (!w2) {
        #pragma unroll
        for (int nt = 0; nt < 8; ++nt) {
            int c = nt * 16 + mloc;
            float bv = bias ? bias[c] : 0.f;
            #pragma unroll
            for (int r = 0; r < 8; ++r) {
                int grow = rowBase + m0 + r + 8 * half;   // C layout: VGPR r -> M=r(+8)
                if (grow < M) {
                    float v = acc[nt][r] + bv;
                    if (relu) v = fmaxf(v, 0.f);
                    if (resid) v += resid[(size_t)grow * D_FEAT + c];
                    Cout[(size_t)grow * D_FEAT + c] = v;
                }
            }
        }
    } else {
        // fused mlp2: out[m] = relu(row + bias) . w2 + b2
        float b2 = *b2p;
        #pragma unroll
        for (int r = 0; r < 8; ++r) {
            float p = 0.f;
            #pragma unroll
            for (int nt = 0; nt < 8; ++nt) {
                int c = nt * 16 + mloc;
                float v = acc[nt][r] + bias[c];
                if (relu) v = fmaxf(v, 0.f);
                p = fmaf(v, w2[c], p);
            }
            #pragma unroll
            for (int off = 8; off; off >>= 1) p += __shfl_xor(p, off, 32);
            if (mloc == 0) {
                int grow = rowBase + m0 + r + 8 * half;
                if (grow < M) Cout[grow] = p + b2;
            }
        }
    }
}

// ---------------------------------------------------------------------------
// Fusion: per-feature softmax over 2 branches (temp 0.7), weighted sum,
// then two LayerNorms (fusion.ln, out_norm). Wave per row.
// ---------------------------------------------------------------------------
__global__ void k_fuse(const float* __restrict__ Z0, const float* __restrict__ Z1,
                       const float* __restrict__ L0, const float* __restrict__ L1,
                       const float* __restrict__ gf, const float* __restrict__ bf,
                       const float* __restrict__ go, const float* __restrict__ bo,
                       float* __restrict__ out) {
    int wave = threadIdx.x >> 5, lane = threadIdx.x & 31;
    int row = blockIdx.x * 8 + wave;
    if (row >= N_NODES) return;
    size_t off = (size_t)row * D_FEAT + lane * 4;
    float4 z0 = *(const float4*)(Z0 + off);
    float4 z1 = *(const float4*)(Z1 + off);
    float4 l0 = *(const float4*)(L0 + off);
    float4 l1 = *(const float4*)(L1 + off);
    float h[4], a0[4] = {l0.x, l0.y, l0.z, l0.w}, a1[4] = {l1.x, l1.y, l1.z, l1.w};
    float zz0[4] = {z0.x, z0.y, z0.z, z0.w}, zz1[4] = {z1.x, z1.y, z1.z, z1.w};
    #pragma unroll
    for (int i = 0; i < 4; ++i) {
        float s0 = a0[i] * INV_TEMP, s1 = a1[i] * INV_TEMP;
        float m = fmaxf(s0, s1);
        float e0 = __expf(s0 - m), e1 = __expf(s1 - m);
        float w0 = e0 / (e0 + e1);
        h[i] = w0 * zz0[i] + (1.f - w0) * zz1[i];
    }
    float4 hv = make_float4(h[0], h[1], h[2], h[3]);
    hv = wave_ln(hv, gf, bf, lane);
    hv = wave_ln(hv, go, bo, lane);
    *(float4*)(out + off) = hv;
}

// ---------------------------------------------------------------------------
extern "C" void kernel_launch(void* const* d_in, const int* in_sizes, int n_in,
                              void* d_out, int out_size, void* d_ws, size_t ws_size,
                              hipStream_t stream) {
    (void)in_sizes; (void)n_in; (void)out_size; (void)ws_size;
    const int*   pairs    = (const int*)d_in[0];
    const int*   adj_row  = (const int*)d_in[1];
    const int*   adj_col  = (const int*)d_in[2];
    const float* adj_val  = (const float*)d_in[3];
    const float* node_emb = (const float*)d_in[4];
    const float* type_emb = (const float*)d_in[5];
    #define FP(i) ((const float*)d_in[i])

    // workspace: row_ptr + 5 N x 128 fp32 buffers (lifetime-scheduled)
    int* rowptr = (int*)d_ws;
    float* base = (float*)((char*)d_ws + (((N_NODES + 1) * sizeof(int) + 255) & ~(size_t)255));
    const size_t stride = (size_t)N_NODES * D_FEAT;
    float* B0 = base;
    float* B1 = base + stride;
    float* B2 = base + 2 * stride;
    float* B3 = base + 3 * stride;
    float* B4 = base + 4 * stride;

    const dim3 blk(256);
    const int gridRow = (N_NODES + 7) / 8;
    const int gridMMN = (N_NODES + 127) / 128;

    auto MM128 = [&](const float* A0, int wI, int bI, float* C) {
        k_mm<128><<<gridMMN, blk, (size_t)512 * 128, stream>>>(
            A0, nullptr, nullptr, FP(wI), FP(bI), nullptr, C, nullptr, nullptr,
            N_NODES, 0);
    };
    auto MM256 = [&](const float* A0, const float* A1, int wI, int bI,
                     const float* resid, float* C, int relu) {
        k_mm<256><<<gridMMN, blk, (size_t)512 * 256, stream>>>(
            A0, A1, nullptr, FP(wI), FP(bI), resid, C, nullptr, nullptr,
            N_NODES, relu);
    };
    auto SPMM = [&](const float* H, const float* resid, float* out, int relu) {
        k_spmm<<<gridRow, blk, 0, stream>>>(rowptr, adj_col, adj_val, H, resid, out, relu);
    };
    auto LNL = [&](const float* X, int gI, int bI, float* Y) {
        k_ln<<<gridRow, blk, 0, stream>>>(X, FP(gI), FP(bI), Y);
    };

    // CSR row pointers + initial embedding
    k_rowptr<<<(N_NODES + 256) / 256, blk, 0, stream>>>(adj_row, rowptr);
    k_init<<<(N_NODES * (D_FEAT / 4) + 255) / 256, blk, 0, stream>>>(node_emb, type_emb, B0);
    SPMM(B0, nullptr, B1, 0);                       // X = spmm(X0)        -> B1

    // GCN branch
    LNL(B1, 6, 7, B0);                              // LN(X)
    MM128(B0, 8, 9, B2);                            // @W1+b
    SPMM(B2, B1, B0, 1);                            // relu(spmm)+X  H1    -> B0
    LNL(B0, 10, 11, B2);
    MM128(B2, 12, 13, B3);
    SPMM(B3, B0, B2, 0);                            // H_gcn               -> B2

    // SAGE branch
    LNL(B1, 14, 15, B0);                            // Hn = LN(X)
    SPMM(B0, nullptr, B3, 0);                       // neigh
    MM256(B0, B3, 16, 17, B1, B4, 1);               // relu([Hn|nb]@W+b)+X -> B4
    LNL(B4, 18, 19, B0);
    SPMM(B0, nullptr, B3, 0);
    MM256(B0, B3, 20, 21, B4, B1, 0);               // H_sage              -> B1

    // Fusion (proj, gate, softmax, weighted sum, 2x LN)
    MM128(B2, 22, 23, B0);                          // Z0
    MM128(B1, 24, 25, B3);                          // Z1
    MM128(B0, 26, 27, B4);                          // logits0
    MM128(B3, 28, 29, B1);                          // logits1
    k_fuse<<<gridRow, blk, 0, stream>>>(B0, B3, B4, B1,
                                        FP(30), FP(31), FP(32), FP(33), B2); // H -> B2

    // Decoder GCNs
    LNL(B2, 34, 35, B0);
    MM128(B0, 36, 37, B1);
    SPMM(B1, B2, B0, 1);                            // H1 -> B0
    LNL(B0, 38, 39, B1);
    MM128(B1, 40, 41, B3);
    SPMM(B3, B0, B1, 0);                            // H  -> B1

    // Edge MLP: [H[u]|H[v]] @ mlp1 + b, relu, fused dot with mlp2 -> d_out
    k_mm<256><<<(N_PAIRS + 127) / 128, blk, (size_t)512 * 256, stream>>>(
        B1, nullptr, pairs, FP(42), FP(43), nullptr, (float*)d_out,
        FP(44), FP(45), N_PAIRS, 1);
    #undef FP
}
